// SparseMoEBlock_88553635709707
// MI455X (gfx1250) — matrix operand order repaired and checked by reading, not compiled, not run on hardware
//
#include <hip/hip_runtime.h>
#include <hip/hip_bf16.h>

// ---------------- problem constants (from reference) ----------------
#define E_ 8
#define A_ 2
#define H_ 2048
#define I_ 5632
#define T_ 8            // B*S tokens
#define TM 16           // WMMA M (tokens padded to 16)
#define TILE_I 128
#define NCHUNK (I_ / TILE_I)   // 44

typedef __bf16 v16bf  __attribute__((ext_vector_type(16)));
typedef __bf16 bf16x8 __attribute__((ext_vector_type(8)));
typedef float  v8f    __attribute__((ext_vector_type(8)));

#define XSTRIDE (H_ + 8)       // bf16 elems; 4112B stride -> rotates LDS banks
#define HSTRIDE (TILE_I + 8)   // 272B stride

// ---------------- router: logits -> softmax -> top2 -> weights ----------------
__global__ void moe_router(const float* __restrict__ x,
                           const float* __restrict__ gate_w,
                           float* __restrict__ out,     // [T*H] main out + [T*A] weights
                           float* __restrict__ wcomb,   // ws: E*16 combine weights
                           int*   __restrict__ flags)   // ws: E active flags
{
    __shared__ float logits[T_ * E_];
    const int tid = threadIdx.x;    // 256 threads

    // zero main output region (atomic-accumulated later)
    for (int j = tid; j < T_ * H_; j += 256) out[j] = 0.0f;

    if (tid < T_ * E_) { // one (t,e) dot product per thread
        const int t = tid >> 3, e = tid & 7;
        const float4* xr = (const float4*)(x + t * H_);
        const float4* gr = (const float4*)(gate_w + e * H_);
        float s = 0.0f;
        for (int k = 0; k < H_ / 4; ++k) {
            float4 a = xr[k], b = gr[k];
            s += a.x * b.x + a.y * b.y + a.z * b.z + a.w * b.w;
        }
        logits[t * E_ + e] = s;
    }
    __syncthreads();
    if (tid < E_ * TM) wcomb[tid] = 0.0f;
    if (tid < E_) flags[tid] = 0;
    __syncthreads();

    if (tid < T_) {
        const int t = tid;
        float l[E_], p[E_], mx = -1e30f;
        for (int e = 0; e < E_; ++e) { l[e] = logits[t * E_ + e]; mx = fmaxf(mx, l[e]); }
        float sum = 0.0f;
        for (int e = 0; e < E_; ++e) { p[e] = __expf(l[e] - mx); sum += p[e]; }
        for (int e = 0; e < E_; ++e) p[e] /= sum;
        int i1 = 0;
        for (int e = 1; e < E_; ++e) if (p[e] > p[i1]) i1 = e;
        int i2 = (i1 == 0) ? 1 : 0;
        for (int e = 0; e < E_; ++e) if (e != i1 && p[e] > p[i2]) i2 = e;
        const float s2 = p[i1] + p[i2];
        const float w1 = p[i1] / s2, w2 = p[i2] / s2;
        out[T_ * H_ + t * A_ + 0] = w1;
        out[T_ * H_ + t * A_ + 1] = w2;
        wcomb[i1 * TM + t] = w1;
        wcomb[i2 * TM + t] = w2;
        flags[i1] = 1;   // benign same-value race
        flags[i2] = 1;
    }
}

// ---------------- fragment helpers (contiguous k order: element j holds k = kb2 + j) ----------------
__device__ __forceinline__ v16bf zero_frag() {
    v16bf z;
#pragma unroll
    for (int j = 0; j < 16; ++j) z[j] = (__bf16)0.0f;
    return z;
}

// A-matrix 16x32 bf16: lane = row (m = lane&15), kb2 = (lane<16)?0:16.
// element j holds K = kb2 + j (contiguous, matches B).
// -> elements 0..15 = K kb2..kb2+15.
__device__ __forceinline__ v16bf a_frag(const __bf16* row, int kbase /* k0+kb2, mult of 16 */) {
    bf16x8 lo = *(const bf16x8*)(row + kbase);
    bf16x8 hi = *(const bf16x8*)(row + kbase + 8);
    v16bf a;
#pragma unroll
    for (int j = 0; j < 8; ++j) { a[j] = lo[j]; a[8 + j] = hi[j]; }
    return a;
}

// B-matrix 32x16 bf16: lane = col (n = lane&15), kb2 = (lane<16)?0:16.
// VGPR v: K = kb2 + 2v, 2v+1 -> element j holds K = kb2 + j.
// p points at W[row=n][k0+kb2], 16 consecutive fp32 (64B, 16B aligned).
__device__ __forceinline__ v16bf b_frag(const float* __restrict__ p) {
    const float4* q = (const float4*)p;
    float4 f0 = q[0], f1 = q[1], f2 = q[2], f3 = q[3];
    v16bf b;
    b[0]  = (__bf16)f0.x; b[1]  = (__bf16)f0.y; b[2]  = (__bf16)f0.z; b[3]  = (__bf16)f0.w;
    b[4]  = (__bf16)f1.x; b[5]  = (__bf16)f1.y; b[6]  = (__bf16)f1.z; b[7]  = (__bf16)f1.w;
    b[8]  = (__bf16)f2.x; b[9]  = (__bf16)f2.y; b[10] = (__bf16)f2.z; b[11] = (__bf16)f2.w;
    b[12] = (__bf16)f3.x; b[13] = (__bf16)f3.y; b[14] = (__bf16)f3.z; b[15] = (__bf16)f3.w;
    return b;
}

// ---------------- main expert kernel ----------------
__global__ __launch_bounds__(256)
void moe_expert_kernel(const float* __restrict__ x,
                       const float* __restrict__ gate_proj,  // [E,I,H]
                       const float* __restrict__ up_proj,    // [E,H,I]
                       const float* __restrict__ down_proj,  // [E,I,H]
                       const float* __restrict__ wcomb,
                       const int*   __restrict__ flags,
                       float* __restrict__ out)
{
    const int e = blockIdx.y;
    if (flags[e] == 0) return;            // uniform: skip inactive expert (saves HBM)
    const int o0 = blockIdx.x * TILE_I;

    __shared__ __align__(16) __bf16 xlds[T_ * XSTRIDE];
    __shared__ __align__(16) __bf16 hlds[TM * HSTRIDE];

    // stage token matrix X (fp32 -> bf16) once per block
    for (int idx = threadIdx.x; idx < T_ * H_; idx += 256) {
        const int t = idx >> 11;           // / H_
        const int k = idx & (H_ - 1);
        xlds[t * XSTRIDE + k] = (__bf16)x[idx];
    }
    __syncthreads();

    const int wave = threadIdx.x >> 5;
    const int lane = threadIdx.x & 31;
    const int n    = lane & 15;                 // col for B/C, row for A
    const int kb   = (lane < 16) ? 0 : 8;       // unused (A also uses kb2)
    const int kb2  = (lane < 16) ? 0 : 16;      // B K sub-base

    // ---- Phase A: G = X*gate^T, D = X*down^T over 16 o-cols per wave ----
    const int n0 = o0 + wave * 16;
    const size_t grow = ((size_t)e * I_ + (size_t)(n0 + n)) * H_;
    const float* gp = gate_proj + grow;
    const float* dp = down_proj + grow;
    const __bf16* xrow = xlds + (n & 7) * XSTRIDE;   // rows 8..15 are logical zeros

    v8f accG, accD;
#pragma unroll
    for (int j = 0; j < 8; ++j) { accG[j] = 0.0f; accD[j] = 0.0f; }

    for (int k0 = 0; k0 < H_; k0 += 64) {
        // stream-ahead prefetch (speculative, 4KB ahead in this lane's rows)
        __builtin_prefetch(gp + k0 + kb2 + 1024, 0, 0);
        __builtin_prefetch(dp + k0 + kb2 + 1024, 0, 0);
#pragma unroll
        for (int u = 0; u < 2; ++u) {
            const int k = k0 + 32 * u;
            v16bf a = a_frag(xrow, k + kb2);
            if (n >= T_) a = zero_frag();       // padded token rows
            v16bf bg = b_frag(gp + k + kb2);
            accG = __builtin_amdgcn_wmma_f32_16x16x32_bf16(false, a, false, bg,
                                                           (short)0, accG, false, false);
            v16bf bd = b_frag(dp + k + kb2);
            accD = __builtin_amdgcn_wmma_f32_16x16x32_bf16(false, a, false, bd,
                                                           (short)0, accD, false, false);
        }
    }

    // h = silu(G) * D -> LDS (C layout: row m = v + 8*(lane>=16), col = n)
    {
        const int mbase = (lane >> 4) << 3;
#pragma unroll
        for (int v = 0; v < 8; ++v) {
            const float g = accG[v];
            const float sig = __builtin_amdgcn_rcpf(1.0f + __expf(-g)); // hw rcp: ok at bf16 noise
            const float h = g * sig * accD[v];
            hlds[(mbase + v) * HSTRIDE + wave * 16 + n] = (__bf16)h;
        }
    }
    __syncthreads();

    // ---- Phase B: y = h * up^T, weighted atomic combine into out ----
    float wrow[8];
#pragma unroll
    for (int v = 0; v < 8; ++v) wrow[v] = wcomb[e * TM + v];

    const __bf16* hrow = hlds + n * HSTRIDE;       // all 16 rows valid (pad rows = 0)
    for (int it = 0; it < 16; ++it) {
        const int i0 = wave * 256 + it * 16;       // 8 waves * 256 = H_
        const float* up = up_proj + ((size_t)e * H_ + (size_t)(i0 + n)) * I_ + o0;
        v8f acc;
#pragma unroll
        for (int j = 0; j < 8; ++j) acc[j] = 0.0f;
#pragma unroll
        for (int kk = 0; kk < TILE_I; kk += 32) {
            // prefetch next i-tile (16 rows ahead), same lane pattern
            __builtin_prefetch(up + kk + kb2 + (size_t)16 * I_, 0, 0);
            v16bf a  = a_frag(hrow, kk + kb2);
            v16bf bu = b_frag(up + kk + kb2);
            acc = __builtin_amdgcn_wmma_f32_16x16x32_bf16(false, a, false, bu,
                                                          (short)0, acc, false, false);
        }
        if (lane < 16) {                            // rows m = 0..7 are real tokens
            const int icol = i0 + n;
#pragma unroll
            for (int v = 0; v < 8; ++v) {
                if (wrow[v] != 0.0f)
                    unsafeAtomicAdd(out + (size_t)v * H_ + icol, wrow[v] * acc[v]);
            }
        }
    }
}

// ---------------- launch ----------------
extern "C" void kernel_launch(void* const* d_in, const int* in_sizes, int n_in,
                              void* d_out, int out_size, void* d_ws, size_t ws_size,
                              hipStream_t stream) {
    const float* x     = (const float*)d_in[0];
    const float* gw    = (const float*)d_in[1];
    const float* gproj = (const float*)d_in[2];
    const float* uproj = (const float*)d_in[3];
    const float* dproj = (const float*)d_in[4];
    float* out   = (float*)d_out;
    float* wcomb = (float*)d_ws;
    int*   flags = (int*)((char*)d_ws + E_ * TM * sizeof(float));

    hipLaunchKernelGGL(moe_router, dim3(1), dim3(256), 0, stream,
                       x, gw, out, wcomb, flags);
    hipLaunchKernelGGL(moe_expert_kernel, dim3(NCHUNK, E_), dim3(256), 0, stream,
                       x, gproj, uproj, dproj, wcomb, flags, out);
}
